// Decoder_54623394071117
// MI455X (gfx1250) — matrix operand order repaired and checked by reading, not compiled, not run on hardware
//
#include <hip/hip_runtime.h>
#include <hip/hip_bf16.h>
#include <math.h>

// ---------------------------------------------------------------------------
// CDNA5 (gfx1250) bf16-WMMA LSTM decoder, round 3.
//   - Weights bf16-prepacked in d_ws (67 MB -> resident in 192 MB L2).
//   - Per step: layer0 -> layer1 -> head (serial recurrence, 512 steps).
//   - Weight slabs streamed global->LDS with global_load_async_to_lds_b128
//     (ASYNCcnt), double-buffered with COMPILE-TIME buffer indices (pipeline
//     unrolled x2 in source) to kill accumulator v_mov churn.
//   - All 8 B fragments of a K-slab loaded into a register array before the
//     8 WMMAs so ds_load_b128s clause and their latency overlaps the math.
// ---------------------------------------------------------------------------

#define Hh   1024
#define Bb   128
#define Vv   64
#define Tt   512
#define G4H  4096          // 4*H
#define UC   32            // units per block -> 64 blocks per layer kernel

typedef __attribute__((ext_vector_type(16))) __bf16 v16bf;
typedef __attribute__((ext_vector_type(8)))  float  v8f;

__device__ __forceinline__ unsigned short f2bf(float f) {
    union { __hip_bfloat16 h; unsigned short u; } cv;
    cv.h = __float2bfloat16(f);
    return cv.u;
}
__device__ __forceinline__ float sigmf(float x) { return 1.0f / (1.0f + expf(-x)); }

// A-matrix 16x32 bf16 fragment: lane M = lane%16,
// halves e=0..15 -> K = k0 + 16*hi + e (contiguous, matches B)
__device__ __forceinline__ v16bf ld_frag_a(const unsigned short* row, int k0, int hi) {
    union { float4 f[2]; v16bf v; } u;
    u.f[0] = *reinterpret_cast<const float4*>(row + k0 + 16 * hi);
    u.f[1] = *reinterpret_cast<const float4*>(row + k0 + 16 * hi + 8);
    return u.v;
}
// B-matrix 32x16 bf16 fragment from an LDS slab laid out [row][32 cols]:
// lane N = lane%16 picks row r, halves e -> K-col = 16*hi + e (32 contiguous B)
__device__ __forceinline__ v16bf ld_frag_b_lds(const unsigned short* sh, int r, int hi) {
    const float4* p = reinterpret_cast<const float4*>(sh + r * 32 + 16 * hi);
    union { float4 f[2]; v16bf v; } u;
    u.f[0] = p[0];
    u.f[1] = p[1];
    return u.v;
}

// Async-copy one K-slab of the block's weight slab into LDS.
// Slab = 128 rows (4 gates x UC units) x 32 K-cols bf16 = 8 KB.
// 512 chunks of 16B over 256 threads -> 2 async b128 per thread (ASYNCcnt +2/wave).
__device__ __forceinline__ void issue_wslab(const unsigned short* __restrict__ wbase,
                                            unsigned short* sh, int u0, int k0,
                                            int Kpitch, int tid) {
    #pragma unroll
    for (int c = 0; c < 2; ++c) {
        int ci = c * 256 + tid;          // chunk id 0..511 (4 chunks per 32-col row)
        int r  = ci >> 2;                // 0..127  (= g*UC + j)
        int c8 = ci & 3;
        int g  = r >> 5, j = r & 31;
        const unsigned short* gp =
            wbase + (size_t)(g * Hh + u0 + j) * Kpitch + k0 + c8 * 8;
        // generic shared pointer: low 32 bits are the LDS byte offset
        unsigned lo = (unsigned)(size_t)(sh + r * 32 + c8 * 8);
        asm volatile("global_load_async_to_lds_b128 %0, %1, off"
                     :: "v"(lo), "v"(gp) : "memory");
    }
}
#define WAIT_ASYNC_LE(n) asm volatile("s_wait_asynccnt " #n ::: "memory")

// One K-slab of MACs: load A frag + all 8 B frags (distinct regs, clause-able),
// then 8 accumulating WMMAs.
__device__ __forceinline__ void compute_slab(v8f acc[8], const unsigned short* arow,
                                             int k0, const unsigned short* sh,
                                             int ml, int hi) {
    v16bf a = ld_frag_a(arow, k0, hi);
    v16bf bf[8];
    #pragma unroll
    for (int g = 0; g < 4; ++g)
        #pragma unroll
        for (int s = 0; s < 2; ++s)
            bf[g * 2 + s] = ld_frag_b_lds(sh, g * UC + s * 16 + ml, hi);
    #pragma unroll
    for (int i = 0; i < 8; ++i)
        acc[i] = __builtin_amdgcn_wmma_f32_16x16x32_bf16(
            false, a, false, bf[i], (short)0, acc[i], false, false);
}

// ---------------------------------------------------------------------------
// Prepack kernels (deterministic, rerun every call)
// ---------------------------------------------------------------------------
__global__ void pack_bf16(const float* __restrict__ src, unsigned short* __restrict__ dst, size_t n) {
    size_t i = (size_t)blockIdx.x * blockDim.x + threadIdx.x;
    size_t st = (size_t)gridDim.x * blockDim.x;
    for (; i < n; i += st) dst[i] = f2bf(src[i]);
}

// Wcat1[d][n][k] : k<2048 -> w_ih1[d][n][k] ; else w_hh1[d][n][k-2048]
__global__ void pack_w1(const float* __restrict__ wih1, const float* __restrict__ whh1,
                        unsigned short* __restrict__ dst) {
    size_t n_tot = (size_t)2 * G4H * 3072;
    size_t i = (size_t)blockIdx.x * blockDim.x + threadIdx.x;
    size_t st = (size_t)gridDim.x * blockDim.x;
    for (; i < n_tot; i += st) {
        size_t dn = i / 3072;          // d*4096 + n
        int    k  = (int)(i % 3072);
        float v = (k < 2048) ? wih1[dn * 2048 + k] : whh1[dn * 1024 + (k - 2048)];
        dst[i] = f2bf(v);
    }
}

// init: l0cat[0], l1cat[0] <- bf16(h0), c-states <- c0, x=MASK_IDX, flag=0
__global__ void init_state(const float* __restrict__ h0, const float* __restrict__ c0in,
                           unsigned short* __restrict__ l0c, unsigned short* __restrict__ l1c,
                           float* __restrict__ c0st, float* __restrict__ c1st,
                           float* __restrict__ xbuf, int* __restrict__ flagbuf) {
    size_t n_tot = (size_t)2 * Bb * Hh;           // per layer
    size_t i = (size_t)blockIdx.x * blockDim.x + threadIdx.x;
    size_t st = (size_t)gridDim.x * blockDim.x;
    for (; i < n_tot; i += st) {
        int d = (int)(i / (Bb * Hh));
        int r = (int)(i % (Bb * Hh));
        int b = r / Hh, u = r % Hh;
        size_t cat = (size_t)b * (2 * Hh) + (size_t)d * Hh + u;
        l0c[cat] = f2bf(h0[(size_t)d * Bb * Hh + r]);          // layer0: dirs 0,1
        l1c[cat] = f2bf(h0[(size_t)(2 + d) * Bb * Hh + r]);    // layer1: dirs 2,3
        c0st[i]  = c0in[(size_t)d * Bb * Hh + r];
        c1st[i]  = c0in[(size_t)(2 + d) * Bb * Hh + r];
        if (i < Bb) { xbuf[i] = 4.0f; flagbuf[i] = 0; }        // MASK_IDX = 4
    }
}

// ---------------------------------------------------------------------------
// Layer 0: gates = h @ W_hh0^T + x*w_ih0 + b0 ; K=1024 (NK=32, even)
// grid = 64 blocks (dir * 32 unit-chunks of UC=32), 256 threads = 8 waves.
// ---------------------------------------------------------------------------
__global__ __launch_bounds__(256) void lstm_layer0(
    const unsigned short* __restrict__ Wcat0,     // (2,4096,1024) bf16
    const unsigned short* __restrict__ l0_read,   // (128,2048) bf16  prev h concat
    unsigned short* __restrict__       l0_write,  // (128,2048) bf16  new h concat
    float* __restrict__                c_state,   // (2,128,1024) f32 (in place)
    const float* __restrict__ w_ih0, const float* __restrict__ bias0,
    const float* __restrict__ xbuf) {
    const int d   = blockIdx.x >> 5;
    const int u0  = (blockIdx.x & 31) * UC;
    const int tid = threadIdx.x;
    const int wv = tid >> 5, ln = tid & 31;
    const int ml = ln & 15, hi = ln >> 4;
    const int m0 = wv * 16;

    __shared__ unsigned short shB[2][128 * 32];   // 2 x 8 KB

    const unsigned short* arow  = l0_read + (size_t)(m0 + ml) * 2048 + (size_t)d * Hh;
    const unsigned short* wbase = Wcat0 + (size_t)d * G4H * Hh;

    v8f acc[8] = {};
    const int NK = Hh / 32;                       // 32, even
    issue_wslab(wbase, shB[0], u0, 0, Hh, tid);
    for (int kk = 0; kk < NK; kk += 2) {
        // ---- slab kk (buf0); prefetch kk+1 (buf1) ----
        issue_wslab(wbase, shB[1], u0, (kk + 1) * 32, Hh, tid);
        WAIT_ASYNC_LE(0x2);
        __syncthreads();
        compute_slab(acc, arow, kk * 32, shB[0], ml, hi);
        __syncthreads();
        // ---- slab kk+1 (buf1); prefetch kk+2 (buf0) ----
        if (kk + 2 < NK) {
            issue_wslab(wbase, shB[0], u0, (kk + 2) * 32, Hh, tid);
            WAIT_ASYNC_LE(0x2);
        } else {
            WAIT_ASYNC_LE(0x0);
        }
        __syncthreads();
        compute_slab(acc, arow, (kk + 1) * 32, shB[1], ml, hi);
        __syncthreads();
    }
    // epilogue: LSTM cell, elementwise across i/f/g/o accumulators
    #pragma unroll
    for (int r = 0; r < 8; ++r) {
        int brow = m0 + r + 8 * hi;
        float xv = xbuf[brow];
        #pragma unroll
        for (int s = 0; s < 2; ++s) {
            int u = u0 + s * 16 + ml;
            const float* wi = w_ih0 + (size_t)d * G4H;
            const float* bz = bias0 + (size_t)d * G4H;
            float gi = acc[0 * 2 + s][r] + xv * wi[u]          + bz[u];
            float gf = acc[1 * 2 + s][r] + xv * wi[Hh + u]     + bz[Hh + u];
            float gg = acc[2 * 2 + s][r] + xv * wi[2 * Hh + u] + bz[2 * Hh + u];
            float go = acc[3 * 2 + s][r] + xv * wi[3 * Hh + u] + bz[3 * Hh + u];
            size_t ci = ((size_t)d * Bb + brow) * Hh + u;
            float c2 = sigmf(gf) * c_state[ci] + sigmf(gi) * tanhf(gg);
            float h2 = sigmf(go) * tanhf(c2);
            c_state[ci] = c2;
            l0_write[(size_t)brow * 2048 + (size_t)d * Hh + u] = f2bf(h2);
        }
    }
}

// ---------------------------------------------------------------------------
// Layer 1: gates = [inp(2H) | h(H)] @ Wcat1^T + b1 ; K=3072 (NK=96, even)
// ---------------------------------------------------------------------------
__global__ __launch_bounds__(256) void lstm_layer1(
    const unsigned short* __restrict__ Wcat1,     // (2,4096,3072) bf16
    const unsigned short* __restrict__ l0_cur,    // (128,2048) layer0 out, this step
    const unsigned short* __restrict__ l1_read,   // (128,2048) prev layer1 h concat
    unsigned short* __restrict__       l1_write,  // (128,2048)
    float* __restrict__ c_state, const float* __restrict__ bias1) {
    const int d   = blockIdx.x >> 5;
    const int u0  = (blockIdx.x & 31) * UC;
    const int tid = threadIdx.x;
    const int wv = tid >> 5, ln = tid & 31;
    const int ml = ln & 15, hi = ln >> 4;
    const int m0 = wv * 16;

    __shared__ unsigned short shB[2][128 * 32];

    const unsigned short* arow0 = l0_cur  + (size_t)(m0 + ml) * 2048;                  // K in [0,2048)
    const unsigned short* arow1 = l1_read + (size_t)(m0 + ml) * 2048 + (size_t)d * Hh; // K in [2048,3072)
    const unsigned short* wbase = Wcat1 + (size_t)d * G4H * 3072;

    v8f acc[8] = {};
    const int NK = 3072 / 32;                     // 96, even
    issue_wslab(wbase, shB[0], u0, 0, 3072, tid);
    for (int kk = 0; kk < NK; kk += 2) {
        int k0a = kk * 32, k0b = (kk + 1) * 32;
        issue_wslab(wbase, shB[1], u0, k0b, 3072, tid);
        WAIT_ASYNC_LE(0x2);
        __syncthreads();
        if (k0a < 2048) compute_slab(acc, arow0, k0a, shB[0], ml, hi);
        else            compute_slab(acc, arow1, k0a - 2048, shB[0], ml, hi);
        __syncthreads();
        if (kk + 2 < NK) {
            issue_wslab(wbase, shB[0], u0, (kk + 2) * 32, 3072, tid);
            WAIT_ASYNC_LE(0x2);
        } else {
            WAIT_ASYNC_LE(0x0);
        }
        __syncthreads();
        if (k0b < 2048) compute_slab(acc, arow0, k0b, shB[1], ml, hi);
        else            compute_slab(acc, arow1, k0b - 2048, shB[1], ml, hi);
        __syncthreads();
    }
    #pragma unroll
    for (int r = 0; r < 8; ++r) {
        int brow = m0 + r + 8 * hi;
        #pragma unroll
        for (int s = 0; s < 2; ++s) {
            int u = u0 + s * 16 + ml;
            const float* bz = bias1 + (size_t)d * G4H;
            float gi = acc[0 * 2 + s][r] + bz[u];
            float gf = acc[1 * 2 + s][r] + bz[Hh + u];
            float gg = acc[2 * 2 + s][r] + bz[2 * Hh + u];
            float go = acc[3 * 2 + s][r] + bz[3 * Hh + u];
            size_t ci = ((size_t)d * Bb + brow) * Hh + u;
            float c2 = sigmf(gf) * c_state[ci] + sigmf(gi) * tanhf(gg);
            float h2 = sigmf(go) * tanhf(c2);
            c_state[ci] = c2;
            l1_write[(size_t)brow * 2048 + (size_t)d * Hh + u] = f2bf(h2);
        }
    }
}

// ---------------------------------------------------------------------------
// Head: logits = l1cat @ lin_w^T + lin_b ; argmax / blank-flag / masked store.
// Single block, 8 waves; tiny (0.4% of FLOPs), plain global B loads.
// ---------------------------------------------------------------------------
__global__ __launch_bounds__(256) void head_kernel(
    const unsigned short* __restrict__ lin_bf,   // (64,2048) bf16
    const unsigned short* __restrict__ l1_cur,   // (128,2048) bf16
    const float* __restrict__ lin_b,
    float* __restrict__ xbuf, int* __restrict__ flagbuf,
    float* __restrict__ out_t, int stride_b) {   // out_t = d_out + t*V ; stride_b = T*V
    __shared__ float logits[Bb][Vv];
    const int wv = threadIdx.x >> 5;
    const int ln = threadIdx.x & 31;
    const int ml = ln & 15, hi = ln >> 4;
    const int m0 = wv * 16;

    const unsigned short* arow = l1_cur + (size_t)(m0 + ml) * 2048;
    v8f acc[4] = {};
    for (int k0 = 0; k0 < 2048; k0 += 32) {
        v16bf a = ld_frag_a(arow, k0, hi);
        v16bf bf[4];
        #pragma unroll
        for (int nt = 0; nt < 4; ++nt) {
            union { float4 f[2]; v16bf v; } ub;
            const float4* p = reinterpret_cast<const float4*>(
                lin_bf + (size_t)(nt * 16 + ml) * 2048 + k0 + 16 * hi);
            ub.f[0] = p[0]; ub.f[1] = p[1];
            bf[nt] = ub.v;
        }
        #pragma unroll
        for (int nt = 0; nt < 4; ++nt)
            acc[nt] = __builtin_amdgcn_wmma_f32_16x16x32_bf16(
                false, a, false, bf[nt], (short)0, acc[nt], false, false);
    }
    #pragma unroll
    for (int r = 0; r < 8; ++r)
        #pragma unroll
        for (int nt = 0; nt < 4; ++nt) {
            int v = nt * 16 + ml;
            logits[m0 + r + 8 * hi][v] = acc[nt][r] + lin_b[v];
        }
    __syncthreads();

    if (threadIdx.x < Bb) {
        int b = threadIdx.x;
        float best = logits[b][0];
        int bi = 0;
        for (int v = 1; v < Vv; ++v) { float t = logits[b][v]; if (t > best) { best = t; bi = v; } }
        int fl = flagbuf[b] | (bi == 1);           // BLANK_IDX = 1
        flagbuf[b] = fl;
        xbuf[b] = (float)bi;                       // argmax (unmasked) feeds next step
        float* orow = out_t + (size_t)b * stride_b;
        for (int v = 0; v < Vv; ++v) {
            float val = logits[b][v];
            if (fl && v != 3) val = 0.0f;          // KEEP_IDX = 3 one-hot mask
            orow[v] = val;
        }
    }
}

// ---------------------------------------------------------------------------
extern "C" void kernel_launch(void* const* d_in, const int* in_sizes, int n_in,
                              void* d_out, int out_size, void* d_ws, size_t ws_size,
                              hipStream_t stream) {
    const float* h0    = (const float*)d_in[0];
    const float* c0    = (const float*)d_in[1];
    const float* w_ih0 = (const float*)d_in[2];
    const float* w_hh0 = (const float*)d_in[3];
    const float* b0    = (const float*)d_in[4];
    const float* w_ih1 = (const float*)d_in[5];
    const float* w_hh1 = (const float*)d_in[6];
    const float* b1    = (const float*)d_in[7];
    const float* lin_w = (const float*)d_in[8];
    const float* lin_b = (const float*)d_in[9];
    float* out = (float*)d_out;

    // workspace carve-up (~72 MB total)
    char* ws = (char*)d_ws;
    size_t off = 0;
    auto carve = [&](size_t bytes) -> char* {
        char* p = ws + off;
        off = (off + bytes + 255) & ~(size_t)255;
        return p;
    };
    unsigned short* Wcat0  = (unsigned short*)carve((size_t)2 * G4H * Hh * 2);    // 16 MB
    unsigned short* Wcat1  = (unsigned short*)carve((size_t)2 * G4H * 3072 * 2);  // 48 MB
    unsigned short* lin_bf = (unsigned short*)carve((size_t)Vv * 2048 * 2);
    unsigned short* l0cat[2], *l1cat[2];
    l0cat[0] = (unsigned short*)carve((size_t)Bb * 2048 * 2);
    l0cat[1] = (unsigned short*)carve((size_t)Bb * 2048 * 2);
    l1cat[0] = (unsigned short*)carve((size_t)Bb * 2048 * 2);
    l1cat[1] = (unsigned short*)carve((size_t)Bb * 2048 * 2);
    float* c0st = (float*)carve((size_t)2 * Bb * Hh * 4);
    float* c1st = (float*)carve((size_t)2 * Bb * Hh * 4);
    float* xbuf = (float*)carve(Bb * 4);
    int*  flagb = (int*)carve(Bb * 4);
    (void)ws_size; (void)in_sizes; (void)n_in; (void)out_size;

    // prepack (deterministic; rerun each call)
    pack_bf16<<<2048, 256, 0, stream>>>(w_hh0, Wcat0, (size_t)2 * G4H * Hh);
    pack_w1  <<<4096, 256, 0, stream>>>(w_ih1, w_hh1, Wcat1);
    pack_bf16<<<128, 256, 0, stream>>>(lin_w, lin_bf, (size_t)Vv * 2048);
    init_state<<<1024, 256, 0, stream>>>(h0, c0, l0cat[0], l1cat[0], c0st, c1st, xbuf, flagb);

    for (int t = 0; t < Tt; ++t) {
        int p = t & 1;
        lstm_layer0<<<64, 256, 0, stream>>>(Wcat0, l0cat[p], l0cat[1 - p], c0st,
                                            w_ih0, b0, xbuf);
        lstm_layer1<<<64, 256, 0, stream>>>(Wcat1, l0cat[1 - p], l1cat[p], l1cat[1 - p],
                                            c1st, b1);
        head_kernel<<<1, 256, 0, stream>>>(lin_bf, l1cat[1 - p], lin_b, xbuf, flagb,
                                           out + (size_t)t * Vv, Tt * Vv);
    }
}
